// _MambaSLModel_24120536334712
// MI455X (gfx1250) — matrix operand order repaired and checked by reading, not compiled, not run on hardware
//
#include <hip/hip_runtime.h>
#include <hip/hip_bf16.h>

// ---------------------------------------------------------------------------
// Model constants
// ---------------------------------------------------------------------------
#define B_    4
#define L_    2048
#define C_IN_ 12
#define DM_   256          // d_model
#define DS_   16           // d_state
#define DCV_  4            // d_conv
#define DI_   512          // d_inner
#define DTR_  16           // dt_rank
#define NCLS_ 10
#define NH_   8
#define KTOK_ 3
#define BL_   (B_ * L_)    // 8192 rows

typedef __attribute__((ext_vector_type(8)))  _Float16 v8h;
typedef __attribute__((ext_vector_type(16))) _Float16 v16h;
typedef __attribute__((ext_vector_type(8)))  float    v8f;
typedef int v4i_ __attribute__((vector_size(16)));   // int4 pointee for async builtin

__device__ __forceinline__ float silu_f(float x) {
    return x / (1.0f + __expf(-x));
}

// ---------------------------------------------------------------------------
// gfx1250 async global->LDS copy (ASYNCcnt path), with safe fallback.
// Builtin signature (from hipcc diagnostic): arg0 = v4i in AS1 (__device__),
// arg1 = v4i in AS3 (__shared__), then imm offset, imm cpol.
// ---------------------------------------------------------------------------
#if defined(__HIP_DEVICE_COMPILE__) &&                                   \
    __has_builtin(__builtin_amdgcn_global_load_async_to_lds_b128) &&      \
    __has_builtin(__builtin_amdgcn_s_wait_asynccnt)
#define ASYNC_LDS_OK 1
#else
#define ASYNC_LDS_OK 0
#endif

__device__ __forceinline__ void cp_b128_g2l(const _Float16* g, _Float16* l) {
#if ASYNC_LDS_OK
    // generic->AS1 is value-preserving; generic LDS pointer's low 32 bits are
    // the LDS byte offset (ISA 10.2).
    __builtin_amdgcn_global_load_async_to_lds_b128(
        (__attribute__((address_space(1))) v4i_*)(unsigned long long)(g),
        (__attribute__((address_space(3))) v4i_*)(unsigned int)(unsigned long long)(l),
        0, 0);
#else
    *(v8h*)l = *(const v8h*)g;
#endif
}

__device__ __forceinline__ void cp_async_wait() {
#if ASYNC_LDS_OK
    __builtin_amdgcn_s_wait_asynccnt(0);
#endif
}

__device__ __forceinline__ v8f wmma_f16(v16h a, v16h b, v8f c) {
    return __builtin_amdgcn_wmma_f32_16x16x32_f16(
        /*neg_a=*/false, a, /*neg_b=*/false, b,
        /*c_mod=*/(short)0, c, /*reuse_a=*/false, /*reuse_b=*/false);
}

// ---------------------------------------------------------------------------
// Generic f32 -> f16 conversion with row-stride + zero padding.
// ---------------------------------------------------------------------------
__global__ void k_cvt_f16_pad(const float* __restrict__ src,
                              _Float16* __restrict__ dst,
                              int Ns, int Ks, int src_stride,
                              int Nd, int Kd) {
    int idx = blockIdx.x * 256 + threadIdx.x;
    if (idx >= Nd * Kd) return;
    int n = idx / Kd;
    int k = idx - n * Kd;
    float v = (n < Ns && k < Ks) ? src[(size_t)n * src_stride + k] : 0.0f;
    dst[idx] = (_Float16)v;
}

// ---------------------------------------------------------------------------
// Token embedding: Conv1d(C_IN->DM, k=3, 'same' edge pad) + positional emb.
// ---------------------------------------------------------------------------
__global__ void k_embed(const float* __restrict__ x_enc,      // [B,L,C_IN]
                        const float* __restrict__ tok_w,      // [DM,C_IN,KTOK]
                        const float* __restrict__ pe,         // [L,DM]
                        _Float16* __restrict__ h16) {         // [BL,DM]
    size_t idx = (size_t)blockIdx.x * 256 + threadIdx.x;
    if (idx >= (size_t)BL_ * DM_) return;
    int d = (int)(idx % DM_);
    size_t bl = idx / DM_;
    int l = (int)(bl % L_);
    int b = (int)(bl / L_);
    float acc = pe[(size_t)l * DM_ + d];
#pragma unroll
    for (int k = 0; k < KTOK_; ++k) {
        int ls = l + k - 1;                                     // pl = 1
        ls = ls < 0 ? 0 : (ls > L_ - 1 ? L_ - 1 : ls);          // replicate edge
        const float* xp = x_enc + ((size_t)b * L_ + ls) * C_IN_;
#pragma unroll
        for (int c = 0; c < C_IN_; ++c)
            acc += xp[c] * tok_w[(d * C_IN_ + c) * KTOK_ + k];
    }
    h16[idx] = (_Float16)acc;
}

// ---------------------------------------------------------------------------
// Simple WMMA GEMM (one 16x16 tile per wave). Used for small / odd N,K GEMMs.
// C[M,N] = A[M,K](f16) * W[N,K]^T(f16), f32 accumulate. K % 32 == 0, N % 16 == 0.
// ---------------------------------------------------------------------------
__global__ void k_gemm_wmma(const _Float16* __restrict__ A,
                            const _Float16* __restrict__ W,
                            float* __restrict__ C,
                            int M, int N, int K) {
    const int lane = threadIdx.x;
    const int wave = threadIdx.y;
    const int n0 = blockIdx.x * 16;
    const int m0 = (blockIdx.y * 8 + wave) * 16;
    if (m0 >= M || n0 >= N) return;      // wave-uniform guard

    const int hi = lane >> 4;
    const int r  = lane & 15;
    const _Float16* Ar = A + (size_t)(m0 + r) * K;
    const _Float16* Wr = W + (size_t)(n0 + r) * K;

    v8f acc = {};
    for (int k0 = 0; k0 < K; k0 += 32) {
        if (k0 + 32 < K) {
            __builtin_prefetch(Ar + k0 + 32, 0, 3);
            __builtin_prefetch(Wr + k0 + 32, 0, 3);
        }
        v8h alo = *(const v8h*)(Ar + k0 + hi * 16);
        v8h ahi = *(const v8h*)(Ar + k0 + hi * 16 + 8);
        v16h bfrag = *(const v16h*)(Wr + k0 + hi * 16);
        v16h afrag;
#pragma unroll
        for (int j = 0; j < 8; ++j) { afrag[j] = alo[j]; afrag[j + 8] = ahi[j]; }
        acc = wmma_f16(afrag, bfrag, acc);
    }
#pragma unroll
    for (int i = 0; i < 8; ++i)
        C[(size_t)(m0 + hi * 8 + i) * N + (n0 + r)] = acc[i];
}

// ---------------------------------------------------------------------------
// Register-blocked WMMA GEMM with async-LDS B staging.
// Block = 8 waves; block tile = 128(M) x 64(N); K slab = 64 (2 wmma k-steps).
// Each wave: 1 A fragment reused across 4 N-tiles -> 4 WMMAs per k-step,
// 8 WMMAs per staged slab. B tile (64 rows x 64 halves) staged once per block
// via global_load_async_to_lds_b128 (ASYNCcnt) and read via ds_load_b128.
// LDS row stride 72 halves (36 dwords): gcd(36,64)=4 -> conflict-free banks.
// Requires N % 64 == 0, K % 64 == 0, M % 128 == 0.
// ---------------------------------------------------------------------------
#define BN_TILE     64
#define BK_SLAB     64
#define LDSB_STRIDE 72   // halves per staged row

__global__ void k_gemm_wmma_lds(const _Float16* __restrict__ A,
                                const _Float16* __restrict__ W,
                                float* __restrict__ C,
                                int M, int N, int K) {
    __shared__ _Float16 Bs[BN_TILE * LDSB_STRIDE];   // 9216 bytes

    const int tid  = threadIdx.y * 32 + threadIdx.x;
    const int lane = threadIdx.x;
    const int wave = threadIdx.y;
    const int n0 = blockIdx.x * BN_TILE;
    const int m0 = (blockIdx.y * 8 + wave) * 16;
    const int hi = lane >> 4;
    const int r  = lane & 15;

    const _Float16* Arow = A + (size_t)(m0 + r) * K;

    // Copy plan: 64 rows x 8 chunks of 16B = 512 chunks; 2 chunks per thread.
    const int c0 = tid, c1 = tid + 256;
    const int row0 = c0 >> 3, col0 = (c0 & 7) * 8;   // halves
    const int row1 = c1 >> 3, col1 = (c1 & 7) * 8;

    v8f acc0 = {}, acc1 = {}, acc2 = {}, acc3 = {};

    for (int k0 = 0; k0 < K; k0 += BK_SLAB) {
        __syncthreads();                              // previous slab consumed
        cp_b128_g2l(W + (size_t)(n0 + row0) * K + k0 + col0,
                    &Bs[row0 * LDSB_STRIDE + col0]);
        cp_b128_g2l(W + (size_t)(n0 + row1) * K + k0 + col1,
                    &Bs[row1 * LDSB_STRIDE + col1]);
        if (k0 + BK_SLAB < K)
            __builtin_prefetch(Arow + k0 + BK_SLAB, 0, 3);
        cp_async_wait();
        __syncthreads();

#pragma unroll
        for (int ks = 0; ks < BK_SLAB; ks += 32) {
            v8h alo = *(const v8h*)(Arow + k0 + ks + hi * 16);
            v8h ahi = *(const v8h*)(Arow + k0 + ks + hi * 16 + 8);
            v16h af;
#pragma unroll
            for (int j = 0; j < 8; ++j) { af[j] = alo[j]; af[j + 8] = ahi[j]; }

            // B fragments for 4 N-tiles from LDS (2 x 16B ds_load each).
            const _Float16* bp = &Bs[r * LDSB_STRIDE + ks + hi * 16];
            v16h b0, b1, b2, b3;
            {
                v8h lo = *(const v8h*)(bp + 0 * 16 * LDSB_STRIDE);
                v8h hi8 = *(const v8h*)(bp + 0 * 16 * LDSB_STRIDE + 8);
#pragma unroll
                for (int j = 0; j < 8; ++j) { b0[j] = lo[j]; b0[j + 8] = hi8[j]; }
            }
            {
                v8h lo = *(const v8h*)(bp + 1 * 16 * LDSB_STRIDE);
                v8h hi8 = *(const v8h*)(bp + 1 * 16 * LDSB_STRIDE + 8);
#pragma unroll
                for (int j = 0; j < 8; ++j) { b1[j] = lo[j]; b1[j + 8] = hi8[j]; }
            }
            {
                v8h lo = *(const v8h*)(bp + 2 * 16 * LDSB_STRIDE);
                v8h hi8 = *(const v8h*)(bp + 2 * 16 * LDSB_STRIDE + 8);
#pragma unroll
                for (int j = 0; j < 8; ++j) { b2[j] = lo[j]; b2[j + 8] = hi8[j]; }
            }
            {
                v8h lo = *(const v8h*)(bp + 3 * 16 * LDSB_STRIDE);
                v8h hi8 = *(const v8h*)(bp + 3 * 16 * LDSB_STRIDE + 8);
#pragma unroll
                for (int j = 0; j < 8; ++j) { b3[j] = lo[j]; b3[j + 8] = hi8[j]; }
            }

            acc0 = wmma_f16(af, b0, acc0);
            acc1 = wmma_f16(af, b1, acc1);
            acc2 = wmma_f16(af, b2, acc2);
            acc3 = wmma_f16(af, b3, acc3);
        }
    }

#pragma unroll
    for (int i = 0; i < 8; ++i) {
        size_t o = (size_t)(m0 + hi * 8 + i) * N + n0 + r;
        C[o]      = acc0[i];
        C[o + 16] = acc1[i];
        C[o + 32] = acc2[i];
        C[o + 48] = acc3[i];
    }
}

// ---------------------------------------------------------------------------
// Causal depthwise conv (k=4, left pad 3) on x-half of xz, + bias, SiLU.
// ---------------------------------------------------------------------------
__global__ void k_dwconv_silu(const float* __restrict__ xz,   // [BL, 2*DI]
                              const float* __restrict__ cw,   // [DI,1,DCV]
                              const float* __restrict__ cb,   // [DI]
                              float* __restrict__ xf,         // [BL, DI]
                              _Float16* __restrict__ x16) {   // [BL, DI]
    size_t idx = (size_t)blockIdx.x * 256 + threadIdx.x;
    if (idx >= (size_t)BL_ * DI_) return;
    int d = (int)(idx % DI_);
    size_t bl = idx / DI_;
    int l = (int)(bl % L_);
    int b = (int)(bl / L_);
    float acc = cb[d];
#pragma unroll
    for (int k = 0; k < DCV_; ++k) {
        int ls = l - (DCV_ - 1) + k;
        if (ls >= 0)
            acc += xz[((size_t)b * L_ + ls) * (2 * DI_) + d] * cw[d * DCV_ + k];
    }
    float sv = silu_f(acc);
    xf[idx]  = sv;
    x16[idx] = (_Float16)sv;
}

// ---------------------------------------------------------------------------
// Selective scan. One thread per (batch, d_inner channel); 16 f32 states in
// registers; sequential over L. Fuses softplus(dt) and silu(z) gating.
// ---------------------------------------------------------------------------
__global__ void k_scan(const float* __restrict__ dtlin,  // [BL, DI]
                       const float* __restrict__ dtb,    // [DI]
                       const float* __restrict__ xf,     // [BL, DI]
                       const float* __restrict__ dbl,    // [BL, 48] dt|B|C
                       const float* __restrict__ xz,     // [BL, 2*DI] (z half)
                       const float* __restrict__ A_log,  // [DI, DS]
                       const float* __restrict__ Dp,     // [DI]
                       float* __restrict__ u,            // [BL, DI]
                       _Float16* __restrict__ u16) {     // [BL, DI]
    int t = blockIdx.x * blockDim.x + threadIdx.x;
    if (t >= B_ * DI_) return;
    int b = t / DI_;
    int d = t % DI_;

    float Ar[DS_], hst[DS_];
#pragma unroll
    for (int n = 0; n < DS_; ++n) {
        Ar[n]  = -__expf(A_log[(size_t)d * DS_ + n]);
        hst[n] = 0.0f;
    }
    const float bias = dtb[d];
    const float Dv   = Dp[d];

    for (int l = 0; l < L_; ++l) {
        size_t row = (size_t)b * L_ + l;
        float dtv = dtlin[row * DI_ + d] + bias;
        dtv = (dtv > 20.0f) ? dtv : __logf(1.0f + __expf(dtv));  // softplus
        float xv = xf[row * DI_ + d];
        const float* dr = dbl + row * (DTR_ + 2 * DS_);
        float acc = 0.0f;
#pragma unroll
        for (int n = 0; n < DS_; ++n) {
            float dA = __expf(dtv * Ar[n]);
            hst[n] = dA * hst[n] + dtv * dr[DTR_ + n] * xv;       // B_t
            acc += hst[n] * dr[DTR_ + DS_ + n];                   // C_t
        }
        float y  = acc + xv * Dv;
        float zv = xz[row * (2 * DI_) + DI_ + d];
        float uu = y * silu_f(zv);
        u[row * DI_ + d]   = uu;
        u16[row * DI_ + d] = (_Float16)uu;
    }
}

// ---------------------------------------------------------------------------
// LayerNorm(d_model) + SiLU. One block (256 threads) per row.
// ---------------------------------------------------------------------------
__global__ void k_ln_silu(const float* __restrict__ y,     // [BL, DM]
                          const float* __restrict__ lnw,
                          const float* __restrict__ lnb,
                          float* __restrict__ mo,          // [BL, DM]
                          _Float16* __restrict__ mo16) {   // [BL, DM]
    int row = blockIdx.x;
    int t = threadIdx.x;
    __shared__ float rs[DM_];
    __shared__ float rs2[DM_];
    float v = y[(size_t)row * DM_ + t];
    rs[t]  = v;
    rs2[t] = v * v;
    __syncthreads();
    for (int st = DM_ / 2; st > 0; st >>= 1) {
        if (t < st) { rs[t] += rs[t + st]; rs2[t] += rs2[t + st]; }
        __syncthreads();
    }
    float mean = rs[0] * (1.0f / DM_);
    float var  = rs2[0] * (1.0f / DM_) - mean * mean;
    float nrm  = (v - mean) * rsqrtf(var + 1e-5f) * lnw[t] + lnb[t];
    float sv   = silu_f(nrm);
    mo[(size_t)row * DM_ + t]   = sv;
    mo16[(size_t)row * DM_ + t] = (_Float16)sv;
}

// ---------------------------------------------------------------------------
// Attention score: s[row] = max_h( mo[row,:] . attn_w[h,:] + attn_b[h] )
// ---------------------------------------------------------------------------
__global__ void k_score(const float* __restrict__ mo,      // [BL, DM]
                        const float* __restrict__ aw,      // [NH, DM]
                        const float* __restrict__ ab,      // [NH]
                        float* __restrict__ s) {           // [BL]
    size_t row = (size_t)blockIdx.x * 256 + threadIdx.x;
    if (row >= BL_) return;
    const float* mr = mo + row * DM_;
    float mx = -3.0e38f;
    for (int h = 0; h < NH_; ++h) {
        float acc = ab[h];
        const float* wr = aw + h * DM_;
        for (int k = 0; k < DM_; ++k) acc += mr[k] * wr[k];
        mx = fmaxf(mx, acc);
    }
    s[row] = mx;
}

// ---------------------------------------------------------------------------
// Softmax-over-L pooling of masked logits. One block per batch.
// ---------------------------------------------------------------------------
__global__ void k_pool(const float* __restrict__ s,       // [BL]
                       const float* __restrict__ logit,   // [BL, 16] (10 valid)
                       const float* __restrict__ mask,    // [BL]
                       float* __restrict__ out) {         // [B, NCLS]
    int b = blockIdx.x;
    int t = threadIdx.x;
    __shared__ float red[256];

    float m = -3.0e38f;
    for (int l = t; l < L_; l += 256) m = fmaxf(m, s[(size_t)b * L_ + l]);
    red[t] = m; __syncthreads();
    for (int st = 128; st > 0; st >>= 1) {
        if (t < st) red[t] = fmaxf(red[t], red[t + st]);
        __syncthreads();
    }
    float mx = red[0]; __syncthreads();

    float sm = 0.0f;
    for (int l = t; l < L_; l += 256) sm += __expf(s[(size_t)b * L_ + l] - mx);
    red[t] = sm; __syncthreads();
    for (int st = 128; st > 0; st >>= 1) {
        if (t < st) red[t] += red[t + st];
        __syncthreads();
    }
    float inv_den = 1.0f / red[0]; __syncthreads();

    float acc[NCLS_];
#pragma unroll
    for (int c = 0; c < NCLS_; ++c) acc[c] = 0.0f;
    for (int l = t; l < L_; l += 256) {
        size_t row = (size_t)b * L_ + l;
        float w  = __expf(s[row] - mx) * inv_den;
        float mk = mask[row];
        const float* lr = logit + row * 16;
#pragma unroll
        for (int c = 0; c < NCLS_; ++c) acc[c] += lr[c] * mk * w;
    }
#pragma unroll
    for (int c = 0; c < NCLS_; ++c) {
        red[t] = acc[c]; __syncthreads();
        for (int st = 128; st > 0; st >>= 1) {
            if (t < st) red[t] += red[t + st];
            __syncthreads();
        }
        if (t == 0) out[b * NCLS_ + c] = red[0];
        __syncthreads();
    }
}

// ---------------------------------------------------------------------------
// Host launcher
// ---------------------------------------------------------------------------
extern "C" void kernel_launch(void* const* d_in, const int* in_sizes, int n_in,
                              void* d_out, int out_size, void* d_ws, size_t ws_size,
                              hipStream_t stream) {
    const float* x_enc      = (const float*)d_in[0];
    const float* x_mark_enc = (const float*)d_in[1];
    const float* tok_conv_w = (const float*)d_in[2];
    const float* pe         = (const float*)d_in[3];
    const float* in_proj_w  = (const float*)d_in[4];
    const float* conv1d_w   = (const float*)d_in[5];
    const float* conv1d_b   = (const float*)d_in[6];
    const float* x_proj_w   = (const float*)d_in[7];
    const float* dt_proj_w  = (const float*)d_in[8];
    const float* dt_proj_b  = (const float*)d_in[9];
    const float* A_log      = (const float*)d_in[10];
    const float* Dp         = (const float*)d_in[11];
    const float* out_proj_w = (const float*)d_in[12];
    const float* ln_w       = (const float*)d_in[13];
    const float* ln_b       = (const float*)d_in[14];
    const float* head_w     = (const float*)d_in[15];
    const float* attn_lin_w = (const float*)d_in[16];
    const float* attn_lin_b = (const float*)d_in[17];
    float* out = (float*)d_out;

    // --- workspace carve-up (256-B aligned slabs) ---
    size_t off = 0;
    auto alloc = [&](size_t bytes) -> void* {
        void* p = (char*)d_ws + off;
        off += (bytes + 255) & ~(size_t)255;
        return p;
    };
    _Float16* h16      = (_Float16*)alloc((size_t)BL_ * DM_ * 2);
    _Float16* w_in16   = (_Float16*)alloc((size_t)(2*DI_) * DM_ * 2);
    float*    xz       = (float*)   alloc((size_t)BL_ * (2*DI_) * 4);
    float*    xf       = (float*)   alloc((size_t)BL_ * DI_ * 4);
    _Float16* x16      = (_Float16*)alloc((size_t)BL_ * DI_ * 2);
    _Float16* w_xp16   = (_Float16*)alloc((size_t)(DTR_+2*DS_) * DI_ * 2);
    float*    dbl      = (float*)   alloc((size_t)BL_ * (DTR_+2*DS_) * 4);
    _Float16* dtin16   = (_Float16*)alloc((size_t)BL_ * 32 * 2);
    _Float16* w_dt16   = (_Float16*)alloc((size_t)DI_ * 32 * 2);
    float*    dtlin    = (float*)   alloc((size_t)BL_ * DI_ * 4);
    float*    u        = (float*)   alloc((size_t)BL_ * DI_ * 4);
    _Float16* u16      = (_Float16*)alloc((size_t)BL_ * DI_ * 2);
    _Float16* w_out16  = (_Float16*)alloc((size_t)DM_ * DI_ * 2);
    float*    yproj    = (float*)   alloc((size_t)BL_ * DM_ * 4);
    float*    mo       = (float*)   alloc((size_t)BL_ * DM_ * 4);
    _Float16* mo16     = (_Float16*)alloc((size_t)BL_ * DM_ * 2);
    _Float16* w_hd16   = (_Float16*)alloc((size_t)16 * DM_ * 2);
    float*    logit    = (float*)   alloc((size_t)BL_ * 16 * 4);
    float*    score    = (float*)   alloc((size_t)BL_ * 4);
    (void)ws_size; (void)in_sizes; (void)n_in; (void)out_size;

    dim3 wblock(32, 8);   // 8 waves of 32 (wave32)

    // --- weight conversions to f16 (with padding where needed) ---
    k_cvt_f16_pad<<<(2*DI_*DM_ + 255)/256, 256, 0, stream>>>(
        in_proj_w, w_in16, 2*DI_, DM_, DM_, 2*DI_, DM_);
    k_cvt_f16_pad<<<((DTR_+2*DS_)*DI_ + 255)/256, 256, 0, stream>>>(
        x_proj_w, w_xp16, DTR_+2*DS_, DI_, DI_, DTR_+2*DS_, DI_);
    k_cvt_f16_pad<<<(DI_*32 + 255)/256, 256, 0, stream>>>(
        dt_proj_w, w_dt16, DI_, DTR_, DTR_, DI_, 32);
    k_cvt_f16_pad<<<(DM_*DI_ + 255)/256, 256, 0, stream>>>(
        out_proj_w, w_out16, DM_, DI_, DI_, DM_, DI_);
    k_cvt_f16_pad<<<(16*DM_ + 255)/256, 256, 0, stream>>>(
        head_w, w_hd16, NCLS_, DM_, DM_, 16, DM_);

    // --- 1. token embedding + pe ---
    k_embed<<<((size_t)BL_*DM_ + 255)/256, 256, 0, stream>>>(
        x_enc, tok_conv_w, pe, h16);

    // --- 2. in_proj GEMM (LDS/async, register-blocked): [8192,256]x[1024,256]^T
    k_gemm_wmma_lds<<<dim3((2*DI_)/BN_TILE, BL_/128), wblock, 0, stream>>>(
        h16, w_in16, xz, BL_, 2*DI_, DM_);

    // --- 3. causal depthwise conv + SiLU ---
    k_dwconv_silu<<<((size_t)BL_*DI_ + 255)/256, 256, 0, stream>>>(
        xz, conv1d_w, conv1d_b, xf, x16);

    // --- 4. x_proj GEMM (simple): [8192,512]x[48,512]^T -> dbl [8192,48] ---
    k_gemm_wmma<<<dim3((DTR_+2*DS_)/16, BL_/128), wblock, 0, stream>>>(
        x16, w_xp16, dbl, BL_, DTR_+2*DS_, DI_);

    // --- 5. dt input: dbl[:, :16] -> f16 padded [8192,32] ---
    k_cvt_f16_pad<<<((size_t)BL_*32 + 255)/256, 256, 0, stream>>>(
        dbl, dtin16, BL_, DTR_, DTR_+2*DS_, BL_, 32);

    // --- 6. dt_proj GEMM (simple, K=32): [8192,32]x[512,32]^T -> dtlin ---
    k_gemm_wmma<<<dim3(DI_/16, BL_/128), wblock, 0, stream>>>(
        dtin16, w_dt16, dtlin, BL_, DI_, 32);

    // --- 7. selective scan (fused softplus + gate) -> u ---
    k_scan<<<(B_*DI_ + 255)/256, 256, 0, stream>>>(
        dtlin, dt_proj_b, xf, dbl, xz, A_log, Dp, u, u16);

    // --- 8. out_proj GEMM (LDS/async): [8192,512]x[256,512]^T -> yproj ---
    k_gemm_wmma_lds<<<dim3(DM_/BN_TILE, BL_/128), wblock, 0, stream>>>(
        u16, w_out16, yproj, BL_, DM_, DI_);

    // --- 9. LayerNorm + SiLU ---
    k_ln_silu<<<BL_, DM_, 0, stream>>>(yproj, ln_w, ln_b, mo, mo16);

    // --- 10. head GEMM (simple): [8192,256]x[16,256]^T -> logit [8192,16] ---
    k_gemm_wmma<<<dim3(1, BL_/128), wblock, 0, stream>>>(
        mo16, w_hd16, logit, BL_, 16, DM_);

    // --- 11. attention scores + softmax pooling ---
    k_score<<<(BL_ + 255)/256, 256, 0, stream>>>(mo, attn_lin_w, attn_lin_b, score);
    k_pool<<<B_, 256, 0, stream>>>(score, logit, x_mark_enc, out);
}
